// LSTM_53300544143476
// MI455X (gfx1250) — matrix operand order repaired and checked by reading, not compiled, not run on hardware
//
#include <hip/hip_runtime.h>

#define DEVI __device__ __forceinline__

typedef __bf16 bf16_t;
typedef bf16_t bf16x8  __attribute__((ext_vector_type(8)));
typedef bf16_t bf16x16 __attribute__((ext_vector_type(16)));
typedef float  f32x8   __attribute__((ext_vector_type(8)));
typedef unsigned int u32x4 __attribute__((ext_vector_type(4)));
typedef unsigned int u32x8 __attribute__((ext_vector_type(8)));
typedef unsigned long long u64;

// Problem dims (from reference)
constexpr int Lz = 2, Bz = 64, Sz = 2048, Ez = 128, Hz = 512, Vz = 64;
constexpr int G4H = 4 * Hz;        // 2048 gate rows
constexpr int K0  = Ez + Hz;       // 640
constexpr int K1  = 2 * Hz;        // 1024
constexpr int HSLICE = 8;          // h-units owned per compute WG
constexpr int NWGC = Hz / HSLICE;  // 64 compute workgroups
constexpr int NB   = NWGC + 1;     // +1 logits workgroup
constexpr int NT   = 128;          // 4 wave32s per workgroup

// LDS layout (compute WG); logits WG reuses offset 0 for outW.
constexpr unsigned OFF_W0S = 0;                        // [32][640] bf16
constexpr unsigned OFF_W1S = 32u * K0 * 2;             // [32][1024] bf16
constexpr unsigned OFF_GST = OFF_W1S + 32u * K1 * 2;   // [64][32] f32
constexpr unsigned OFF_XS  = OFF_GST + 64u * 32 * 4;   // 2 x [64][128] bf16
constexpr unsigned XTILE_B = (unsigned)Bz * Ez * 2;    // 16 KB per x tile
constexpr unsigned SMEM_SZ = OFF_XS + 2u * XTILE_B;    // 147456 B = 144 KB

union AB16 { bf16x16 v; bf16x8 h[2]; };

DEVI f32x8 wmma_bf16(bf16x16 a, bf16x16 b, f32x8 c) {
  // 8 args: (neg_a, A, neg_b, B, c_mod, C, reuse_a, reuse_b)
  return __builtin_amdgcn_wmma_f32_16x16x32_bf16(false, a, false, b, (short)0, c,
                                                 false, false);
}

DEVI bf16x8 ld8(const bf16_t* p) { return *(const bf16x8*)p; }

// B-matrix (KxN) fragment from an LDS-resident weight slice stored row-major
// as (N rows x Krow cols). Lane L holds column n = L%16, K-ranges
// kbase + (L/16)*8 .. +8 and kbase + 16 + (L/16)*8 .. +8 (table order).
DEVI bf16x16 loadB_lds(const bf16_t* wsl, int Krow, int nt, int kbase, int lane) {
  int n  = nt * 16 + (lane & 15);
  int kb = kbase + ((lane >> 4) << 3);
  const bf16_t* p = wsl + (size_t)n * (size_t)Krow + kb;
  AB16 u; u.h[0] = ld8(p); u.h[1] = ld8(p + 16);
  return u.v;
}

// ---- Tensor Data Mover: build D# (groups 0/1) and issue tensor_load_to_lds.
DEVI void desc_put(unsigned* w, int lo, int width, u64 v) {
  int hi = lo + width;
  for (int d = lo >> 5; (d << 5) < hi; ++d) {
    int s  = d << 5;
    int b0 = lo > s ? lo : s;
    int b1 = hi < s + 32 ? hi : s + 32;
    u64 mask = (b1 - b0 >= 64) ? ~0ull : ((1ull << (b1 - b0)) - 1ull);
    u64 part = (v >> (b0 - lo)) & mask;
    w[d] |= (unsigned)(part << (b0 - s));
  }
}

union G0u { unsigned w[4]; u32x4 v; };
union G1u { unsigned w[8]; u32x8 v; };

// 2-D tile load: rows x cols elements (2 bytes each), global row stride in
// elements, packed into LDS at byte offset lds_off (relative to LDS base).
DEVI void tdm_load_2d(unsigned lds_off, u64 gaddr, unsigned rows, unsigned cols,
                      u64 row_stride_elems) {
  G0u a; G1u b;
#pragma unroll
  for (int i = 0; i < 4; ++i) a.w[i] = 0u;
#pragma unroll
  for (int i = 0; i < 8; ++i) b.w[i] = 0u;
  a.w[0] = 1u;                                        // count=1 (valid user D#)
  a.w[1] = lds_off;                                   // lds_addr [63:32]
  a.w[2] = (unsigned)gaddr;                           // global_addr[31:0]
  a.w[3] = (unsigned)((gaddr >> 32) & 0x01FFFFFFull)  // global_addr[56:32]
         | (2u << 30);                                // type=2 ("image")
  b.w[0] = 1u << 16;                                  // data_size=1 -> 2 bytes
  desc_put(b.w, 48, 32, cols);                        // tensor_dim0
  desc_put(b.w, 80, 32, rows);                        // tensor_dim1
  desc_put(b.w, 112, 16, cols);                       // tile_dim0
  desc_put(b.w, 128, 16, rows);                       // tile_dim1
  desc_put(b.w, 160, 48, row_stride_elems);           // tensor_dim0_stride
  asm volatile("tensor_load_to_lds %0, %1" :: "s"(a.v), "s"(b.v) : "memory");
}

// Grid-wide arrive + generation barrier in workspace memory.
DEVI void gbar(unsigned* cnt, volatile unsigned* gen, unsigned expected, unsigned my_gen) {
  __syncthreads();
  if (threadIdx.x == 0) {
    __threadfence();
    unsigned old = atomicAdd(cnt, 1u);
    if (old == expected - 1u) {
      atomicSub(cnt, expected);
      __threadfence();
      atomicAdd((unsigned*)gen, 1u);
    }
    while (*gen < my_gen) __builtin_amdgcn_s_sleep(2);
  }
  __syncthreads();
}

DEVI float sigm(float x) { return 1.0f / (1.0f + __expf(-x)); }

// ---------------------------------------------------------------------------
// Prep: embedding gather + bf16 weight casts + h seed + barrier reset.
// ---------------------------------------------------------------------------
__global__ void lstm_prep(const int* __restrict__ x, const float* __restrict__ h0,
                          const float* __restrict__ emb,
                          const float* __restrict__ W0, const float* __restrict__ W1,
                          const float* __restrict__ outW,
                          bf16_t* __restrict__ xemb, bf16_t* __restrict__ w0b,
                          bf16_t* __restrict__ w1b, bf16_t* __restrict__ owb,
                          bf16_t* __restrict__ h0bf, bf16_t* __restrict__ h1bf,
                          unsigned* __restrict__ bar) {
  size_t idx    = (size_t)blockIdx.x * blockDim.x + threadIdx.x;
  size_t stride = (size_t)gridDim.x * blockDim.x;
  if (idx == 0) { bar[0] = 0u; bar[1] = 0u; }
  const size_t nE  = (size_t)Bz * Sz * Ez;
  const size_t nW0 = (size_t)G4H * K0;
  const size_t nW1 = (size_t)G4H * K1;
  const size_t nOW = (size_t)Vz * Hz;
  const size_t nH  = (size_t)Bz * Hz;
  const size_t total = nE + nW0 + nW1 + nOW + 2 * nH;
  for (size_t i = idx; i < total; i += stride) {
    if (i < nE) {
      size_t e = i % Ez, bt = i / Ez;               // bt = b*S + t
      int tok = x[bt];
      xemb[i] = (bf16_t)emb[(size_t)tok * Ez + e];
    } else if (i < nE + nW0) {
      size_t j = i - nE; w0b[j] = (bf16_t)W0[j];
    } else if (i < nE + nW0 + nW1) {
      size_t j = i - nE - nW0; w1b[j] = (bf16_t)W1[j];
    } else if (i < nE + nW0 + nW1 + nOW) {
      size_t j = i - nE - nW0 - nW1; owb[j] = (bf16_t)outW[j];
    } else if (i < nE + nW0 + nW1 + nOW + nH) {
      size_t j = i - nE - nW0 - nW1 - nOW;          // seed parity-1 h (read at t=0)
      h0bf[nH + j] = (bf16_t)h0[j];
    } else {
      size_t j = i - nE - nW0 - nW1 - nOW - nH;
      h1bf[nH + j] = (bf16_t)h0[nH + j];
    }
  }
}

// ---------------------------------------------------------------------------
// Persistent LSTM: 64 compute WGs (gate slices, TDM-staged LDS weights,
// TDM-prefetched x tiles) + 1 logits WG (TDM-staged outW, one step behind).
// ---------------------------------------------------------------------------
__global__ void lstm_main(const float* __restrict__ c0in,
                          const float* __restrict__ b0v,
                          const float* __restrict__ b1v,
                          const float* __restrict__ outb,
                          const bf16_t* __restrict__ xemb,
                          const bf16_t* __restrict__ w0b,
                          const bf16_t* __restrict__ w1b,
                          const bf16_t* __restrict__ owb,
                          bf16_t* __restrict__ h0bf,   // 2 x B*H (parity)
                          bf16_t* __restrict__ h1bf,   // 2 x B*H (parity)
                          unsigned* __restrict__ bar,
                          float* __restrict__ dout) {
  extern __shared__ __attribute__((aligned(16))) char smem[];
  const int tid = threadIdx.x, lane = tid & 31, wave = tid >> 5;
  unsigned* cnt = bar;
  volatile unsigned* gen = bar + 1;
  const size_t nH = (size_t)Bz * Hz;
  const size_t logits_sz = (size_t)Bz * Sz * Vz;
  const f32x8 z8 = {0.f, 0.f, 0.f, 0.f, 0.f, 0.f, 0.f, 0.f};

  const int ml    = wave * 16 + (lane & 15);  // batch row for this lane's A frag
  const int khalf = (lane >> 4) * 8;          // A-layout: lanes0-15 K 0..7/16..23

  if (blockIdx.x < NWGC) {
    const int wg = blockIdx.x;
    bf16_t* w0s = (bf16_t*)(smem + OFF_W0S);
    bf16_t* w1s = (bf16_t*)(smem + OFF_W1S);
    float*  gst = (float*)(smem + OFF_GST);

    // ---- TDM-stage weight slices (rows packed [f8|i8|g8|o8]) + x tile t=0.
    if (wave == 0) {
#pragma unroll
      for (int g = 0; g < 4; ++g) {
        tdm_load_2d(OFF_W0S + (unsigned)g * 8u * K0 * 2,
                    (u64)(const void*)(w0b + (size_t)(g * Hz + wg * HSLICE) * K0),
                    8, K0, (u64)K0);
        tdm_load_2d(OFF_W1S + (unsigned)g * 8u * K1 * 2,
                    (u64)(const void*)(w1b + (size_t)(g * Hz + wg * HSLICE) * K1),
                    8, K1, (u64)K1);
      }
      // x tile for t=0: rows=batch, row stride S*E elements
      tdm_load_2d(OFF_XS, (u64)(const void*)xemb, Bz, Ez, (u64)Sz * Ez);
    }
    __builtin_amdgcn_s_wait_tensorcnt(0);
    __syncthreads();

    float creg[2][4];   // cell state (layer, local pair) — lives in registers
    unsigned mygen = 0;

    for (int t = 0; t < Sz; ++t) {
      // ======================= layer 0 =======================
      // x[t] tile is in LDS (TDM). Kick off async prefetch of x[t+1].
      __builtin_amdgcn_s_wait_tensorcnt(0);
      __syncthreads();
      if (wave == 0 && t + 1 < Sz)
        tdm_load_2d(OFF_XS + ((unsigned)(t + 1) & 1u) * XTILE_B,
                    (u64)(const void*)(xemb + (size_t)(t + 1) * Ez),
                    Bz, Ez, (u64)Sz * Ez);
      const bf16_t* xs = (const bf16_t*)(smem + OFF_XS + ((unsigned)t & 1u) * XTILE_B);

      const bf16_t* h0prev = h0bf + (size_t)((t + 1) & 1) * nH;
      f32x8 acc0 = z8, acc1 = z8;
      for (int kc = 0; kc < K0 / 32; ++kc) {
        int kb  = kc * 32;
        int ka  = kb + khalf;
        int ka2 = kb + 16 + khalf;
        AB16 a;
        const bf16_t* p0 = (ka  < Ez) ? (xs + (size_t)ml * Ez + ka)
                                      : (h0prev + (size_t)ml * Hz + (ka - Ez));
        const bf16_t* p1 = (ka2 < Ez) ? (xs + (size_t)ml * Ez + ka2)
                                      : (h0prev + (size_t)ml * Hz + (ka2 - Ez));
        a.h[0] = ld8(p0);
        a.h[1] = ld8(p1);
        acc0 = wmma_bf16(a.v, loadB_lds(w0s, K0, 0, kb, lane), acc0);
        acc1 = wmma_bf16(a.v, loadB_lds(w0s, K0, 1, kb, lane), acc1);
      }
      {  // stage raw gates to LDS (C layout: col=lane%16, rows=(lane/16)*8+r)
        int col0 = lane & 15, rowb = wave * 16 + (lane >> 4) * 8;
#pragma unroll
        for (int r = 0; r < 8; ++r) {
          gst[(rowb + r) * 32 + col0]      = acc0[r];
          gst[(rowb + r) * 32 + 16 + col0] = acc1[r];
        }
      }
      __syncthreads();
      bf16_t* h0cur = h0bf + (size_t)(t & 1) * nH;
#pragma unroll
      for (int p = 0; p < 4; ++p) {
        int e = tid * 4 + p, b = e >> 3, j = e & 7, hidx = wg * HSLICE + j;
        float pf = gst[b * 32 + j]      + b0v[0 * Hz + hidx];
        float pi = gst[b * 32 + 8 + j]  + b0v[1 * Hz + hidx];
        float pg = gst[b * 32 + 16 + j] + b0v[2 * Hz + hidx];
        float po = gst[b * 32 + 24 + j] + b0v[3 * Hz + hidx];
        float cp = (t == 0) ? c0in[(size_t)b * Hz + hidx] : creg[0][p];
        float cn = sigm(pf) * cp + sigm(pi) * tanhf(pg);
        float hn = sigm(po) * tanhf(cn);
        creg[0][p] = cn;
        h0cur[(size_t)b * Hz + hidx] = (bf16_t)hn;
        if (t == Sz - 1) {
          dout[logits_sz + (size_t)b * Hz + hidx]                       = hn;
          dout[logits_sz + (size_t)Lz * nH + (size_t)b * Hz + hidx]     = cn;
        }
      }
      gbar(cnt, gen, NB, ++mygen);

      // ======================= layer 1 =======================
      const bf16_t* h0c    = h0bf + (size_t)(t & 1) * nH;        // written this step
      const bf16_t* h1prev = h1bf + (size_t)((t + 1) & 1) * nH;
      acc0 = z8; acc1 = z8;
      for (int kc = 0; kc < K1 / 32; ++kc) {
        int kb  = kc * 32;
        int ka  = kb + khalf;
        int ka2 = kb + 16 + khalf;
        AB16 a;
        const bf16_t* p0 = (ka  < Hz) ? (h0c + (size_t)ml * Hz + ka)
                                      : (h1prev + (size_t)ml * Hz + (ka - Hz));
        const bf16_t* p1 = (ka2 < Hz) ? (h0c + (size_t)ml * Hz + ka2)
                                      : (h1prev + (size_t)ml * Hz + (ka2 - Hz));
        a.h[0] = ld8(p0);
        a.h[1] = ld8(p1);
        acc0 = wmma_bf16(a.v, loadB_lds(w1s, K1, 0, kb, lane), acc0);
        acc1 = wmma_bf16(a.v, loadB_lds(w1s, K1, 1, kb, lane), acc1);
      }
      {
        int col0 = lane & 15, rowb = wave * 16 + (lane >> 4) * 8;
#pragma unroll
        for (int r = 0; r < 8; ++r) {
          gst[(rowb + r) * 32 + col0]      = acc0[r];
          gst[(rowb + r) * 32 + 16 + col0] = acc1[r];
        }
      }
      __syncthreads();
      bf16_t* h1cur = h1bf + (size_t)(t & 1) * nH;
#pragma unroll
      for (int p = 0; p < 4; ++p) {
        int e = tid * 4 + p, b = e >> 3, j = e & 7, hidx = wg * HSLICE + j;
        float pf = gst[b * 32 + j]      + b1v[0 * Hz + hidx];
        float pi = gst[b * 32 + 8 + j]  + b1v[1 * Hz + hidx];
        float pg = gst[b * 32 + 16 + j] + b1v[2 * Hz + hidx];
        float po = gst[b * 32 + 24 + j] + b1v[3 * Hz + hidx];
        float cp = (t == 0) ? c0in[nH + (size_t)b * Hz + hidx] : creg[1][p];
        float cn = sigm(pf) * cp + sigm(pi) * tanhf(pg);
        float hn = sigm(po) * tanhf(cn);
        creg[1][p] = cn;
        h1cur[(size_t)b * Hz + hidx] = (bf16_t)hn;
        if (t == Sz - 1) {
          dout[logits_sz + nH + (size_t)b * Hz + hidx]                   = hn;
          dout[logits_sz + (size_t)Lz * nH + nH + (size_t)b * Hz + hidx] = cn;
        }
      }
      gbar(cnt, gen, NB, ++mygen);
    }
  } else {
    // ======================= logits workgroup =======================
    bf16_t* ows = (bf16_t*)smem;  // [64][512] outW row-major, LDS-resident
    if (wave == 0)  // single contiguous 64x512 block -> one TDM descriptor
      tdm_load_2d(0u, (u64)(const void*)owb, 1, (unsigned)(Vz * Hz),
                  (u64)(Vz * Hz));
    __builtin_amdgcn_s_wait_tensorcnt(0);
    __syncthreads();
    unsigned mygen = 0;

    auto logits_step = [&](int t) {
      const bf16_t* h1b = h1bf + (size_t)(t & 1) * nH;
      f32x8 acc[4] = {z8, z8, z8, z8};
      for (int kc = 0; kc < Hz / 32; ++kc) {
        int kb = kc * 32;
        AB16 a;
        a.h[0] = ld8(h1b + (size_t)ml * Hz + kb + khalf);
        a.h[1] = ld8(h1b + (size_t)ml * Hz + kb + 16 + khalf);
#pragma unroll
        for (int nt = 0; nt < 4; ++nt)
          acc[nt] = wmma_bf16(a.v, loadB_lds(ows, Hz, nt, kb, lane), acc[nt]);
      }
#pragma unroll
      for (int nt = 0; nt < 4; ++nt) {
        int v = nt * 16 + (lane & 15);
        float bo = outb[v];
        int rowb = wave * 16 + (lane >> 4) * 8;
#pragma unroll
        for (int r = 0; r < 8; ++r)
          dout[((size_t)(rowb + r) * Sz + t) * Vz + v] = acc[nt][r] + bo;
      }
    };

    for (int t = 0; t < Sz; ++t) {
      if (t > 0) logits_step(t - 1);   // overlaps with layer0 of step t
      gbar(cnt, gen, NB, ++mygen);     // phase A barrier
      gbar(cnt, gen, NB, ++mygen);     // phase B barrier
    }
    logits_step(Sz - 1);
  }
}

// ---------------------------------------------------------------------------
extern "C" void kernel_launch(void* const* d_in, const int* in_sizes, int n_in,
                              void* d_out, int out_size, void* d_ws, size_t ws_size,
                              hipStream_t stream) {
  (void)in_sizes; (void)n_in; (void)out_size; (void)ws_size;
  const int*   x    = (const int*)  d_in[0];
  const float* h0   = (const float*)d_in[1];
  const float* c0   = (const float*)d_in[2];
  const float* emb  = (const float*)d_in[3];
  const float* W0   = (const float*)d_in[4];
  const float* b0   = (const float*)d_in[5];
  const float* W1   = (const float*)d_in[6];
  const float* b1   = (const float*)d_in[7];
  const float* outW = (const float*)d_in[8];
  const float* outb = (const float*)d_in[9];
  float* dout = (float*)d_out;

  char* ws = (char*)d_ws;
  size_t off = 0;
  auto alloc = [&](size_t bytes) {
    size_t o = off;
    off = (off + bytes + 255) & ~(size_t)255;
    return o;
  };
  size_t o_bar  = alloc(256);
  size_t o_xemb = alloc((size_t)Bz * Sz * Ez * 2);
  size_t o_w0   = alloc((size_t)G4H * K0 * 2);
  size_t o_w1   = alloc((size_t)G4H * K1 * 2);
  size_t o_ow   = alloc((size_t)Vz * Hz * 2);
  size_t o_h0   = alloc((size_t)2 * Bz * Hz * 2);
  size_t o_h1   = alloc((size_t)2 * Bz * Hz * 2);

  bf16_t*   xemb = (bf16_t*)(ws + o_xemb);
  bf16_t*   w0b  = (bf16_t*)(ws + o_w0);
  bf16_t*   w1b  = (bf16_t*)(ws + o_w1);
  bf16_t*   owb  = (bf16_t*)(ws + o_ow);
  bf16_t*   h0bf = (bf16_t*)(ws + o_h0);
  bf16_t*   h1bf = (bf16_t*)(ws + o_h1);
  unsigned* bar  = (unsigned*)(ws + o_bar);

  lstm_prep<<<2048, 256, 0, stream>>>(x, h0, emb, W0, W1, outW,
                                      xemb, w0b, w1b, owb, h0bf, h1bf, bar);

  lstm_main<<<NB, NT, SMEM_SZ, stream>>>(c0, b0, b1, outb,
                                         xemb, w0b, w1b, owb,
                                         h0bf, h1bf, bar, dout);
}
